// Decoder_73194832658812
// MI455X (gfx1250) — matrix operand order repaired and checked by reading, not compiled, not run on hardware
//
#include <hip/hip_runtime.h>
#include <math.h>

// ---------------------------------------------------------------------------
// MI455X (gfx1250, wave32) decoder step.
// Heavy GEMMs run through v_wmma_f32_16x16x32_bf16. The dominant stage
// (enc[32768x2048] @ lin2_w^T[2048x1024], 137 GFLOP) stages the activation
// tile in LDS as bf16 and fuses tanh(h_x + h_m + b)*v so the 134 MB h_m
// tensor never touches memory. v3: fp32->bf16 pair-pack via v_perm_b32
// (3 VALU / 2 elements) since cvt_pk_bf16_f32 builtin is absent here.
// ---------------------------------------------------------------------------

typedef __attribute__((ext_vector_type(16))) __bf16        v16bf;
typedef __attribute__((ext_vector_type(8)))  float         v8f;
typedef __attribute__((ext_vector_type(8)))  unsigned int  v8u;

#define B_   64
#define S_   512
#define H_   1024
#define E_   512
#define V_   32000
#define DH_  2048          // D*H (memory dim)
#define XK_  2560          // DH + E (GRU input dim)

// ---- fp32 -> packed bf16x2 -------------------------------------------------
#if __has_builtin(__builtin_amdgcn_cvt_pk_bf16_f32)
__device__ __forceinline__ unsigned pk2(float lo, float hi) {
  return __builtin_bit_cast(unsigned, __builtin_amdgcn_cvt_pk_bf16_f32(lo, hi));
}
#elif __has_builtin(__builtin_amdgcn_perm)
// round-half-up then byte-permute the two high halves into one DWORD:
// v_add_nc_u32 x2 + v_perm_b32 = 3 VALU per 2 elements.
__device__ __forceinline__ unsigned pk2(float lo, float hi) {
  unsigned ul = __builtin_bit_cast(unsigned, lo) + 0x8000u;
  unsigned uh = __builtin_bit_cast(unsigned, hi) + 0x8000u;
  return __builtin_amdgcn_perm(uh, ul, 0x07060302u);   // {uh[31:16], ul[31:16]}
}
#else
__device__ __forceinline__ unsigned short f2bf(float f) {
  union { float f; unsigned u; } v; v.f = f;
  unsigned r = v.u + 0x7FFFu + ((v.u >> 16) & 1u);
  return (unsigned short)(r >> 16);
}
__device__ __forceinline__ unsigned pk2(float lo, float hi) {
  return ((unsigned)f2bf(hi) << 16) | (unsigned)f2bf(lo);
}
#endif

#if __has_builtin(__builtin_amdgcn_tanhf)
#define TANHF __builtin_amdgcn_tanhf
#elif __has_builtin(__builtin_amdgcn_tanh_f32)
#define TANHF __builtin_amdgcn_tanh_f32
#else
#define TANHF tanhf
#endif

__device__ __forceinline__ v8f wmma_bf16(v8u a, v8u b, v8f c) {
  return __builtin_amdgcn_wmma_f32_16x16x32_bf16(
      false, __builtin_bit_cast(v16bf, a),
      false, __builtin_bit_cast(v16bf, b),
      (short)0, c, false, false);
}

// One 16x16 f32-accum tile, K-loop in steps of 32, fp32 sources pair-packed to
// bf16 in-register. Abase = &A[m0*lda] (row-major). Brow = &Bmat[n*ldb] for
// this lane's output column n (weights row-major over K -> contiguous loads).
__device__ __forceinline__ v8f gemm_f32(const float* __restrict__ Abase, int lda,
                                        const float* __restrict__ Brow,
                                        int K, int lane, v8f acc) {
  const int m  = lane & 15;
  const int hi = lane >> 4;
  const float* ap = Abase + m * lda + (hi << 3);   // A frag: K g0..g0+7, g0+16..+23
  const float* bp = Brow + (hi << 3);              // B frag: K g0..g0+7, g0+16..+23
  for (int kb = 0; kb < K; kb += 32) {
    float4 a0 = *(const float4*)(ap + kb);
    float4 a1 = *(const float4*)(ap + kb + 4);
    float4 a2 = *(const float4*)(ap + kb + 16);
    float4 a3 = *(const float4*)(ap + kb + 20);
    float4 b0 = *(const float4*)(bp + kb);
    float4 b1 = *(const float4*)(bp + kb + 4);
    float4 b2 = *(const float4*)(bp + kb + 16);
    float4 b3 = *(const float4*)(bp + kb + 20);
    v8u af, bf;
    af[0] = pk2(a0.x, a0.y); af[1] = pk2(a0.z, a0.w);
    af[2] = pk2(a1.x, a1.y); af[3] = pk2(a1.z, a1.w);
    af[4] = pk2(a2.x, a2.y); af[5] = pk2(a2.z, a2.w);
    af[6] = pk2(a3.x, a3.y); af[7] = pk2(a3.z, a3.w);
    bf[0] = pk2(b0.x, b0.y); bf[1] = pk2(b0.z, b0.w);
    bf[2] = pk2(b1.x, b1.y); bf[3] = pk2(b1.z, b1.w);
    bf[4] = pk2(b2.x, b2.y); bf[5] = pk2(b2.z, b2.w);
    bf[6] = pk2(b3.x, b3.y); bf[7] = pk2(b3.z, b3.w);
    acc = wmma_bf16(af, bf, acc);
  }
  return acc;
}

// --------------------------- tiny prep kernels -----------------------------

// lin2_w fp32 -> bf16 (re-read ~500x from L2 by k_energy; convert once).
__global__ void k_cvt_lin2(const float* __restrict__ w, unsigned short* __restrict__ o) {
  int i = (blockIdx.x * 256 + threadIdx.x) * 4;
  float4 v = *(const float4*)(w + i);
  uint2 pv; pv.x = pk2(v.x, v.y); pv.y = pk2(v.z, v.w);
  *(uint2*)(o + i) = pv;                 // global_store_b64
}

// word embedding gather into x[:, 2048:2560]
__global__ void k_embed(const int* __restrict__ word, const float* __restrict__ emb,
                        float* __restrict__ x) {
  int b = blockIdx.x;
  int w = word[b];
  for (int e = threadIdx.x; e < E_; e += 256)
    x[b * XK_ + DH_ + e] = emb[(size_t)w * E_ + e];
}

// h_x = hidden @ lin1_w^T + lin1_b   [64 x 1024]
__global__ void k_hx(const float* __restrict__ hidden, const float* __restrict__ w1,
                     const float* __restrict__ b1, float* __restrict__ hx) {
  int lane = threadIdx.x & 31;
  int wid  = blockIdx.x * 8 + (threadIdx.x >> 5);   // 256 waves total
  int mt = wid & 3, nt = wid >> 2;                  // 4 m-tiles x 64 n-tiles
  int n  = nt * 16 + (lane & 15);
  v8f z = {};
  v8f acc = gemm_f32(hidden + mt * 16 * H_, H_, w1 + (size_t)n * H_, H_, lane, z);
  float bias = b1[n];
  int hi = lane >> 4;
#pragma unroll
  for (int r = 0; r < 8; ++r)
    hx[(mt * 16 + hi * 8 + r) * H_ + n] = acc[r] + bias;
}

// ------------------- fused attention-energy kernel (dominant) --------------
// e[b,s] = sum_h v[h] * tanh(h_x[b,h] + lin2_b[h] + (enc[b,s,:] . lin2_w[h,:])) + v_b
#define BT_M 64
#define KDP  (DH_ + 8)     // padded LDS row stride (ushorts) -> no bank aliasing

__global__ void __launch_bounds__(256) k_energy(
    const float* __restrict__ enc, const unsigned short* __restrict__ w2bf,
    const float* __restrict__ hx, const float* __restrict__ b2,
    const float* __restrict__ vw, const float* __restrict__ vb,
    float* __restrict__ eout) {
  extern __shared__ unsigned short smem[];
  unsigned short* alds  = smem;                          // 64 x KDP bf16
  float*          epart = (float*)(smem + BT_M * KDP);   // 64 partial energies
  const int b  = blockIdx.x >> 3;
  const int s0 = (blockIdx.x & 7) * BT_M;
  const int tid = threadIdx.x;

  // Stage 64 x 2048 activation tile -> bf16 LDS (enc read exactly once here).
  const float* abase = enc + (size_t)(b * S_ + s0) * DH_;
  for (int i = tid; i < BT_M * (DH_ / 4); i += 256) {
    int row = i >> 9;                 // DH_/4 == 512
    int col = (i & 511) * 4;
    float4 v = *(const float4*)(abase + (size_t)row * DH_ + col);
    uint2 pv; pv.x = pk2(v.x, v.y); pv.y = pk2(v.z, v.w);
    *(uint2*)(alds + row * KDP + col) = pv;      // ds_store_b64
  }
  if (tid < BT_M) epart[tid] = 0.0f;
  __syncthreads();

  const int lane = tid & 31, widx = tid >> 5;
  const int m = lane & 15, hi = lane >> 4;
  float part[4][8];
#pragma unroll
  for (int s = 0; s < 4; ++s)
#pragma unroll
    for (int r = 0; r < 8; ++r) part[s][r] = 0.0f;

  // 2-way N-blocking: each A fragment (ds_load_b128 x2) feeds 2 WMMAs; each
  // pair of B fragments feeds 8 WMMAs across 4 M-subtiles.
  for (int p = widx; p < 32; p += 8) {
    const int n0 = (p * 2) * 16 + m;
    const int n1 = n0 + 16;
    const unsigned short* bp0 = w2bf + (size_t)n0 * DH_ + (hi << 4);
    const unsigned short* bp1 = bp0 + (size_t)16 * DH_;
    v8f acc0[4], acc1[4];
#pragma unroll
    for (int s = 0; s < 4; ++s) { acc0[s] = (v8f){}; acc1[s] = (v8f){}; }
    for (int kb = 0; kb < DH_; kb += 32) {
      uint4 u0 = *(const uint4*)(bp0 + kb);
      uint4 u1 = *(const uint4*)(bp0 + kb + 8);
      uint4 w0 = *(const uint4*)(bp1 + kb);
      uint4 w1 = *(const uint4*)(bp1 + kb + 8);
      v8u bf0, bf1;
      bf0[0] = u0.x; bf0[1] = u0.y; bf0[2] = u0.z; bf0[3] = u0.w;
      bf0[4] = u1.x; bf0[5] = u1.y; bf0[6] = u1.z; bf0[7] = u1.w;
      bf1[0] = w0.x; bf1[1] = w0.y; bf1[2] = w0.z; bf1[3] = w0.w;
      bf1[4] = w1.x; bf1[5] = w1.y; bf1[6] = w1.z; bf1[7] = w1.w;
#pragma unroll
      for (int sub = 0; sub < 4; ++sub) {
        const unsigned short* ap = alds + (sub * 16 + m) * KDP + (hi << 4) + kb;
        uint4 x0 = *(const uint4*)(ap);          // ds_load_b128
        uint4 x1 = *(const uint4*)(ap + 8);
        v8u af;
        af[0] = x0.x; af[1] = x0.y; af[2] = x0.z; af[3] = x0.w;
        af[4] = x1.x; af[5] = x1.y; af[6] = x1.z; af[7] = x1.w;
        acc0[sub] = wmma_bf16(af, bf0, acc0[sub]);
        acc1[sub] = wmma_bf16(af, bf1, acc1[sub]);
      }
    }
    // Fused epilogue: never materialize h_m.
    const float hx0 = hx[b * H_ + n0], bb0 = b2[n0], vv0 = vw[n0];
    const float hx1 = hx[b * H_ + n1], bb1 = b2[n1], vv1 = vw[n1];
#pragma unroll
    for (int sub = 0; sub < 4; ++sub)
#pragma unroll
      for (int r = 0; r < 8; ++r) {
        part[sub][r] += TANHF(acc0[sub][r] + hx0 + bb0) * vv0;
        part[sub][r] += TANHF(acc1[sub][r] + hx1 + bb1) * vv1;
      }
  }
#pragma unroll
  for (int sub = 0; sub < 4; ++sub)
#pragma unroll
    for (int r = 0; r < 8; ++r)
      atomicAdd(&epart[sub * 16 + hi * 8 + r], part[sub][r]);   // ds_add_f32
  __syncthreads();
  if (tid < BT_M) eout[b * S_ + s0 + tid] = epart[tid] + vb[0];
}

// ------------------------------ softmax over S -----------------------------
__global__ void k_softmax(const float* __restrict__ e, float* __restrict__ score) {
  __shared__ float red[256];
  int b = blockIdx.x, t = threadIdx.x;
  float v0 = e[b * S_ + t], v1 = e[b * S_ + 256 + t];
  red[t] = fmaxf(v0, v1); __syncthreads();
  for (int o = 128; o > 0; o >>= 1) {
    if (t < o) red[t] = fmaxf(red[t], red[t + o]);
    __syncthreads();
  }
  float mx = red[0]; __syncthreads();
  float ex0 = __expf(v0 - mx), ex1 = __expf(v1 - mx);
  red[t] = ex0 + ex1; __syncthreads();
  for (int o = 128; o > 0; o >>= 1) {
    if (t < o) red[t] += red[t + o];
    __syncthreads();
  }
  float inv = 1.0f / red[0];
  score[b * S_ + t] = ex0 * inv;
  score[b * S_ + 256 + t] = ex1 * inv;
}

// ---------------------- context = score . enc  -> x[:, :2048] --------------
__global__ void k_context(const float* __restrict__ enc, const float* __restrict__ score,
                          float* __restrict__ x) {
  int b    = blockIdx.x >> 3;
  int mcol = (blockIdx.x & 7) * 256 + threadIdx.x;
  const float* ep = enc + (size_t)b * S_ * DH_ + mcol;
  const float* sp = score + b * S_;
  float acc = 0.0f;
  for (int s = 0; s < S_; ++s) acc += sp[s] * ep[(size_t)s * DH_];
  x[b * XK_ + mcol] = acc;
}

// ------------------------------- GRU cell ----------------------------------
__global__ void k_gru(const float* __restrict__ x, const float* __restrict__ hidden,
                      const float* __restrict__ Wih, const float* __restrict__ bih,
                      const float* __restrict__ Whh, const float* __restrict__ bhh,
                      float* __restrict__ newh) {
  int lane = threadIdx.x & 31;
  int wid  = blockIdx.x * 8 + (threadIdx.x >> 5);   // 256 waves
  int mt = wid & 3, nt = wid >> 2;
  int m0 = mt * 16;
  int n  = nt * 16 + (lane & 15);
  int hi = lane >> 4;
  const float* Ax = x + m0 * XK_;
  const float* Ah = hidden + m0 * H_;
  v8f z = {};
  v8f ir  = gemm_f32(Ax, XK_, Wih + (size_t)(0 * H_ + n) * XK_, XK_, lane, z);
  v8f hr  = gemm_f32(Ah, H_,  Whh + (size_t)(0 * H_ + n) * H_,  H_,  lane, z);
  v8f iz  = gemm_f32(Ax, XK_, Wih + (size_t)(1 * H_ + n) * XK_, XK_, lane, z);
  v8f hz  = gemm_f32(Ah, H_,  Whh + (size_t)(1 * H_ + n) * H_,  H_,  lane, z);
  v8f in_ = gemm_f32(Ax, XK_, Wih + (size_t)(2 * H_ + n) * XK_, XK_, lane, z);
  v8f hn  = gemm_f32(Ah, H_,  Whh + (size_t)(2 * H_ + n) * H_,  H_,  lane, z);
  float bir = bih[n],          bhr = bhh[n];
  float biz = bih[H_ + n],     bhz = bhh[H_ + n];
  float bin = bih[2 * H_ + n], bhn = bhh[2 * H_ + n];
#pragma unroll
  for (int r = 0; r < 8; ++r) {
    int row = m0 + hi * 8 + r;
    float hv = hidden[row * H_ + n];
    float rg = 1.0f / (1.0f + __expf(-(ir[r] + bir + hr[r] + bhr)));
    float zg = 1.0f / (1.0f + __expf(-(iz[r] + biz + hz[r] + bhz)));
    float ng = TANHF(in_[r] + bin + rg * (hn[r] + bhn));
    newh[row * H_ + n] = (1.0f - zg) * ng + zg * hv;
  }
}

// --------------------------- output projection -----------------------------
__global__ void k_out(const float* __restrict__ newh, const float* __restrict__ Wout,
                      const float* __restrict__ bout, float* __restrict__ pred) {
  int lane = threadIdx.x & 31;
  int wid  = blockIdx.x * 8 + (threadIdx.x >> 5);   // 8000 waves
  int mt = wid & 3, nt = wid >> 2;                  // nt 0..1999
  int n  = nt * 16 + (lane & 15);
  v8f z = {};
  v8f acc = gemm_f32(newh + (mt * 16) * H_, H_, Wout + (size_t)n * H_, H_, lane, z);
  float bias = bout[n];
  int hi = lane >> 4;
#pragma unroll
  for (int r = 0; r < 8; ++r)
    pred[(size_t)(mt * 16 + hi * 8 + r) * V_ + n] = acc[r] + bias;
}

// ---------------------------------------------------------------------------
extern "C" void kernel_launch(void* const* d_in, const int* in_sizes, int n_in,
                              void* d_out, int out_size, void* d_ws, size_t ws_size,
                              hipStream_t stream) {
  (void)in_sizes; (void)n_in; (void)out_size; (void)ws_size;
  const int*   word   = (const int*)  d_in[0];
  const float* hidden = (const float*)d_in[1];
  const float* enc    = (const float*)d_in[2];
  // d_in[3] output_distribution, d_in[4] encoder_attention: unused by reference math
  const float* emb    = (const float*)d_in[5];
  const float* l1w    = (const float*)d_in[6];
  const float* l1b    = (const float*)d_in[7];
  const float* l2w    = (const float*)d_in[8];
  const float* l2b    = (const float*)d_in[9];
  const float* vw     = (const float*)d_in[10];
  const float* vb     = (const float*)d_in[11];
  const float* Wih    = (const float*)d_in[12];
  const float* bih    = (const float*)d_in[13];
  const float* Whh    = (const float*)d_in[14];
  const float* bhh    = (const float*)d_in[15];
  const float* Wout   = (const float*)d_in[16];
  const float* bout   = (const float*)d_in[17];

  float* out   = (float*)d_out;                      // pred | new_hidden | score
  float* pred  = out;
  float* newh  = out + (size_t)B_ * V_;
  float* score = out + (size_t)B_ * V_ + (size_t)B_ * H_;

  // workspace: hx[64x1024] | e[64x512] | x[64x2560] | lin2_w bf16 (~5.2 MB total)
  float* ws   = (float*)d_ws;
  float* hx   = ws;
  float* elog = ws + B_ * H_;
  float* x    = ws + B_ * H_ + B_ * S_;
  unsigned short* l2bf = (unsigned short*)(ws + B_ * H_ + B_ * S_ + B_ * XK_);

  k_cvt_lin2<<<(H_ * DH_) / 1024, 256, 0, stream>>>(l2w, l2bf);
  k_embed  <<<B_, 256, 0, stream>>>(word, emb, x);
  k_hx     <<<32, 256, 0, stream>>>(hidden, l1w, l1b, hx);

  size_t shB = (size_t)BT_M * KDP * sizeof(unsigned short) + BT_M * sizeof(float);
  k_energy <<<(B_ * S_) / BT_M, 256, shB, stream>>>(enc, l2bf, hx, l2b, vw, vb, elog);

  k_softmax<<<B_, 256, 0, stream>>>(elog, score);
  k_context<<<B_ * 8, 256, 0, stream>>>(enc, score, x);
  k_gru    <<<32, 256, 0, stream>>>(x, hidden, Wih, bih, Whh, bhh, newh);
  k_out    <<<1000, 256, 0, stream>>>(newh, Wout, bout, pred);
}
